// StockCorrelationGNN_15925738734304
// MI455X (gfx1250) — compile-verified, matrix operand order checked by reading, not run on hardware
//
#include <hip/hip_runtime.h>
#include <stdint.h>

// ---------------- problem constants ----------------
#define NN 10000          // nodes
#define NE 320000         // edges (before self loops)
#define NET (NE + NN)     // edges incl. self loops
#define FN 10
#define FE 6
#define HIDC 128
#define NHEADS 4
#define OUTC 64
#define EPS_BN 1e-5f

typedef _Float16 f16;
typedef _Float16 v16h __attribute__((ext_vector_type(16)));
typedef float    v8f  __attribute__((ext_vector_type(8)));
typedef int      vi4  __attribute__((ext_vector_type(4)));

#if defined(__has_builtin)
#if __has_builtin(__builtin_amdgcn_global_load_async_to_lds_b128)
#define HAVE_ASYNC_LDS 1
#endif
#if __has_builtin(__builtin_amdgcn_s_wait_asynccnt)
#define HAVE_WAIT_ASYNC_BUILTIN 1
#endif
#endif
#ifndef HAVE_ASYNC_LDS
#define HAVE_ASYNC_LDS 0
#endif
#ifndef HAVE_WAIT_ASYNC_BUILTIN
#define HAVE_WAIT_ASYNC_BUILTIN 0
#endif

__device__ inline void wait_async0() {
#if HAVE_ASYNC_LDS
#if HAVE_WAIT_ASYNC_BUILTIN
  __builtin_amdgcn_s_wait_asynccnt(0);
#else
  asm volatile("s_wait_asynccnt 0x0" ::: "memory");
#endif
#endif
}

static inline int cdiv(long a, long b) { return (int)((a + b - 1) / b); }

// monotone float -> uint key (for atomicMax-based segment max)
__device__ inline unsigned fkey(float f) {
  unsigned u = __float_as_uint(f);
  return (u & 0x80000000u) ? ~u : (u | 0x80000000u);
}
__device__ inline float fkey_dec(unsigned k) {
  unsigned u = (k & 0x80000000u) ? (k ^ 0x80000000u) : ~k;
  return __uint_as_float(u);
}

// ---------------- generic small kernels ----------------
__global__ void k_fill(float* p, float v, long n) {
  long i = (long)blockIdx.x * blockDim.x + threadIdx.x;
  if (i < n) p[i] = v;
}

__global__ void k_f32_to_f16(const float* __restrict__ s, f16* __restrict__ d, long n) {
  long i = (long)blockIdx.x * blockDim.x + threadIdx.x;
  if (i < n) d[i] = (f16)s[i];
}

// column means of edge_attr [NE,6] -> cm (sums; divided later)
__global__ void k_colmean_accum(const float* __restrict__ ea, float* __restrict__ cm) {
  __shared__ float s6[FE];
  int tid = threadIdx.x;
  if (tid < FE) s6[tid] = 0.f;
  __syncthreads();
  float loc[FE];
#pragma unroll
  for (int f = 0; f < FE; f++) loc[f] = 0.f;
  for (long e = (long)blockIdx.x * blockDim.x + tid; e < NE; e += (long)gridDim.x * blockDim.x) {
#pragma unroll
    for (int f = 0; f < FE; f++) loc[f] += ea[e * FE + f];
  }
#pragma unroll
  for (int f = 0; f < FE; f++) atomicAdd(&s6[f], loc[f]);
  __syncthreads();
  if (tid < FE) atomicAdd(&cm[tid], s6[tid]);
}
__global__ void k_cm_final(float* cm) {
  int t = threadIdx.x;
  if (t < FE) cm[t] *= (1.0f / (float)NE);
}

// node encoder matmul (K=10, bias omitted: cancels under BN)
__global__ void k_enc_mm(const float* __restrict__ x, const float* __restrict__ w,
                         float* __restrict__ z) {
  long i = (long)blockIdx.x * blockDim.x + threadIdx.x;
  if (i >= (long)NN * HIDC) return;
  int n = (int)(i / HIDC), c = (int)(i % HIDC);
  float s = 0.f;
#pragma unroll
  for (int f = 0; f < FN; f++) s += x[n * FN + f] * w[f * HIDC + c];
  z[i] = s;
}

// BN stats: one block per channel
__global__ void k_bn_stats(const float* __restrict__ z, float* __restrict__ stats,
                           int n, int C) {
  int ch = blockIdx.x, tid = threadIdx.x;
  float s = 0.f, s2 = 0.f;
  for (int i = tid; i < n; i += 256) {
    float v = z[(long)i * C + ch];
    s += v; s2 += v * v;
  }
  __shared__ float ss[256], sq[256];
  ss[tid] = s; sq[tid] = s2;
  __syncthreads();
  for (int o = 128; o > 0; o >>= 1) {
    if (tid < o) { ss[tid] += ss[tid + o]; sq[tid] += sq[tid + o]; }
    __syncthreads();
  }
  if (tid == 0) {
    float m = ss[0] / n;
    stats[ch] = m;
    stats[C + ch] = sq[0] / n - m * m;
  }
}

// BN apply + activation (+ optional f16 copy). act: 0=ReLU, 1=ELU
__global__ void k_bn_apply(const float* __restrict__ z, const float* __restrict__ stats,
                           const float* __restrict__ g, const float* __restrict__ beta,
                           float* __restrict__ outf, f16* __restrict__ outh,
                           int n, int C, int act) {
  long i = (long)blockIdx.x * blockDim.x + threadIdx.x;
  if (i >= (long)n * C) return;
  int c = (int)(i % C);
  float m = stats[c], v = stats[C + c];
  float y = g[c] * (z[i] - m) * rsqrtf(v + EPS_BN) + beta[c];
  y = (act == 0) ? fmaxf(y, 0.f) : (y > 0.f ? y : expf(y) - 1.f);
  outf[i] = y;
  if (outh) outh[i] = (f16)y;
}

// ---------------- WMMA f16 GEMM: C[M,N] = A[M,K] * B[K,N] ----------------
#define GPAD 40
template <int K, int N>
__global__ __launch_bounds__(256) void k_gemm_f16(const f16* __restrict__ A,
                                                  const f16* __restrict__ B,
                                                  float* __restrict__ C, int M) {
  __shared__ f16 As[64 * GPAD];  // [m][k]
  __shared__ f16 Bt[64 * GPAD];  // [n][k] (transposed)
  int tid = threadIdx.x;
  int bm0 = blockIdx.x * 64, bn0 = blockIdx.y * 64;
  int lane = tid & 31, wave = tid >> 5;
  int r = lane & 15, hi = lane >> 4;
  int nsub = wave & 3, msb = (wave >> 2) << 1;   // wave owns rows [msb*16, msb*16+32)
  v8f acc0 = {}; v8f acc1 = {};

  int arow = tid >> 2, akc = (tid & 3) * 8;      // A tile: 64 rows x 32 k
  int brow = tid >> 3, bnc = (tid & 7) * 8;      // B tile: 32 k x 64 n

  for (int k0 = 0; k0 < K; k0 += 32) {
    int gm = bm0 + arow;
#if HAVE_ASYNC_LDS
    // direct global -> LDS async copy (ASYNCcnt), skips the register file
    if (gm < M) {
      __builtin_amdgcn_global_load_async_to_lds_b128(
          (__attribute__((address_space(1))) vi4*)(A + (long)gm * K + k0 + akc),
          (__attribute__((address_space(3))) vi4*)(&As[arow * GPAD + akc]),
          0, 0);
    } else {
#pragma unroll
      for (int i = 0; i < 8; i++) As[arow * GPAD + akc + i] = (f16)0.f;
    }
#else
    {
      union { int4 v; f16 h[8]; } ta;
      if (gm < M) ta.v = *(const int4*)(A + (long)gm * K + k0 + akc);
      else        ta.v = make_int4(0, 0, 0, 0);
#pragma unroll
      for (int i = 0; i < 8; i++) As[arow * GPAD + akc + i] = ta.h[i];
    }
#endif
    // B tile: load row-major, store transposed [n][k] into LDS
    {
      union { int4 v; f16 h[8]; } tb;
      tb.v = *(const int4*)(B + (long)(k0 + brow) * N + bn0 + bnc);
#pragma unroll
      for (int i = 0; i < 8; i++) Bt[(bnc + i) * GPAD + brow] = tb.h[i];
    }
    // prefetch next K-step of B into L2 (global_prefetch_b8)
    if (k0 + 32 < K) __builtin_prefetch(B + (long)(k0 + 32 + brow) * N + bn0 + bnc, 0, 1);
    wait_async0();
    __syncthreads();

    v16h af0, af1, bf;
    int mr0 = (msb * 16 + r) * GPAD;
    int mr1 = ((msb + 1) * 16 + r) * GPAD;
    int ncr = (nsub * 16 + r) * GPAD;
#pragma unroll
    for (int j = 0; j < 8; j++) {
      int kb = (j < 4) ? (hi * 8 + 2 * j) : (16 + hi * 8 + 2 * (j - 4));
      af0[2 * j] = As[mr0 + kb]; af0[2 * j + 1] = As[mr0 + kb + 1];
      af1[2 * j] = As[mr1 + kb]; af1[2 * j + 1] = As[mr1 + kb + 1];
      bf [2 * j] = Bt[ncr + kb]; bf [2 * j + 1] = Bt[ncr + kb + 1];
    }
    acc0 = __builtin_amdgcn_wmma_f32_16x16x32_f16(false, af0, false, bf, (short)0, acc0, false, false);
    acc1 = __builtin_amdgcn_wmma_f32_16x16x32_f16(false, af1, false, bf, (short)0, acc1, false, false);
    __syncthreads();
  }
  int col = bn0 + nsub * 16 + r;
#pragma unroll
  for (int j = 0; j < 8; j++) {
    int g0 = bm0 + msb * 16 + hi * 8 + j;
    int g1 = g0 + 16;
    if (g0 < M) C[(long)g0 * N + col] = acc0[j];
    if (g1 < M) C[(long)g1 * N + col] = acc1[j];
  }
}

// ---------------- attention pipeline ----------------
// per-node src/dst scores from xs
__global__ void k_node_scores(const float* __restrict__ xs, const float* __restrict__ as_,
                              const float* __restrict__ ad_, float* __restrict__ sS,
                              float* __restrict__ sD, int n, int H, int C) {
  int i = blockIdx.x * blockDim.x + threadIdx.x;
  if (i >= n * H) return;
  int nn = i / H, hh = i % H;
  const float* row = xs + (long)nn * H * C + hh * C;
  const float* a = as_ + hh * C;
  const float* d = ad_ + hh * C;
  float s1 = 0.f, s2 = 0.f;
  for (int c = 0; c < C; c++) { s1 += row[c] * a[c]; s2 += row[c] * d[c]; }
  sS[i] = s1; sD[i] = s2;
}

// ce[k,h] = sum_c lew[k, h*C+c] * ae[h,c]
__global__ void k_ce(const float* __restrict__ lew, const float* __restrict__ ae,
                     float* __restrict__ ce, int H, int C) {
  int i = blockIdx.x * blockDim.x + threadIdx.x;
  if (i >= HIDC * H) return;
  int k = i / H, hh = i % H;
  float s = 0.f;
  for (int c = 0; c < C; c++) s += lew[(long)k * H * C + hh * C + c] * ae[hh * C + c];
  ce[i] = s;
}

// edvec[f,h] = edge_w[f,:] @ ce[:,h]; ebias[h] = edge_b @ ce[:,h];
// selfs[h] = (colmean @ edge_w + edge_b) @ ce[:,h]
__global__ void k_edge_vec(const float* __restrict__ ew, const float* __restrict__ eb,
                           const float* __restrict__ cm, const float* __restrict__ ce,
                           float* __restrict__ edvec, float* __restrict__ ebias,
                           float* __restrict__ selfs, int H) {
  int t = threadIdx.x;
  if (t < FE * H) {
    int f = t / H, hh = t % H;
    float s = 0.f;
    for (int k = 0; k < HIDC; k++) s += ew[f * HIDC + k] * ce[k * H + hh];
    edvec[t] = s;
  }
  __syncthreads();
  if (t < H) {
    float s = 0.f;
    for (int k = 0; k < HIDC; k++) s += eb[k] * ce[k * H + t];
    float sf = s;
    for (int f = 0; f < FE; f++) sf += cm[f] * edvec[f * H + t];
    ebias[t] = s;
    selfs[t] = sf;
  }
}

// pass 1: raw scores + leaky relu + segment max (ordered-uint atomicMax)
__global__ void k_edge_pass1(const int* __restrict__ ei, const float* __restrict__ eattr,
                             const float* __restrict__ edvec, const float* __restrict__ ebias,
                             const float* __restrict__ selfs, const float* __restrict__ sS,
                             const float* __restrict__ sD, float* __restrict__ alpha,
                             unsigned* __restrict__ mx, int H) {
  int e = blockIdx.x * blockDim.x + threadIdx.x;
  if (e >= NET) return;
  int s, d;
  float ev[FE];
  if (e < NE) {
    s = ei[e]; d = ei[NE + e];
#pragma unroll
    for (int f = 0; f < FE; f++) ev[f] = eattr[(long)e * FE + f];
  } else { s = d = e - NE; }
  for (int hh = 0; hh < H; hh++) {
    float es;
    if (e < NE) {
      es = ebias[hh];
#pragma unroll
      for (int f = 0; f < FE; f++) es += ev[f] * edvec[f * H + hh];
    } else es = selfs[hh];
    float a = sS[s * H + hh] + sD[d * H + hh] + es;
    a = (a > 0.f) ? a : 0.2f * a;
    alpha[(long)e * H + hh] = a;
    atomicMax(&mx[d * H + hh], fkey(a));
  }
}

// pass 2: exp(raw - max), denom accumulation
__global__ void k_edge_pass2(const int* __restrict__ ei, float* __restrict__ alpha,
                             const unsigned* __restrict__ mx, float* __restrict__ dn, int H) {
  int e = blockIdx.x * blockDim.x + threadIdx.x;
  if (e >= NET) return;
  int d = (e < NE) ? ei[NE + e] : (e - NE);
  for (int hh = 0; hh < H; hh++) {
    float m = fkey_dec(mx[d * H + hh]);
    float v = expf(alpha[(long)e * H + hh] - m);
    alpha[(long)e * H + hh] = v;
    atomicAdd(&dn[d * H + hh], v);
  }
}

// pass 3: weighted scatter-add of xs[src] into out[dst] (block per edge)
__global__ __launch_bounds__(128) void k_agg(const int* __restrict__ ei,
                                             const float* __restrict__ alpha,
                                             const float* __restrict__ dn,
                                             const float* __restrict__ xs,
                                             float* __restrict__ out, int H, int C) {
  int e = blockIdx.x, tid = threadIdx.x;
  int s, d;
  if (e < NE) { s = ei[e]; d = ei[NE + e]; } else { s = d = e - NE; }
  int HC = H * C;
  const float* xr = xs + (long)s * HC;
  float* orow = out + (long)d * HC;
  for (int c = tid; c < HC; c += 128) {
    int hh = c / C;
    float w = alpha[(long)e * H + hh] / (dn[d * H + hh] + 1e-16f);
    atomicAdd(&orow[c], w * xr[c]);
  }
}

// ---------------- readout ----------------
__global__ void k_pool(const float* __restrict__ h, float* __restrict__ ge) {
  int c = blockIdx.x, tid = threadIdx.x;
  float s = 0.f, m = -3.0e38f;
  for (int i = tid; i < NN; i += 256) {
    float v = h[(long)i * OUTC + c];
    s += v; m = fmaxf(m, v);
  }
  __shared__ float ss[256], sm[256];
  ss[tid] = s; sm[tid] = m;
  __syncthreads();
  for (int o = 128; o > 0; o >>= 1) {
    if (tid < o) { ss[tid] += ss[tid + o]; sm[tid] = fmaxf(sm[tid], sm[tid + o]); }
    __syncthreads();
  }
  if (tid == 0) { ge[c] = ss[0] / NN; ge[OUTC + c] = sm[0]; }
}

__global__ void k_mlp(const float* __restrict__ ge, const float* __restrict__ w1,
                      const float* __restrict__ b1, const float* __restrict__ w2,
                      const float* __restrict__ b2, float* __restrict__ out) {
  __shared__ float gg[128], hid[128];
  int t = threadIdx.x;
  gg[t] = ge[t];
  __syncthreads();
  float s = b1[t];
  for (int i = 0; i < 128; i++) s += gg[i] * w1[i * 128 + t];
  hid[t] = fmaxf(s, 0.f);
  __syncthreads();
  if (t < 64) {
    float o = b2[t];
    for (int j = 0; j < 128; j++) o += hid[j] * w2[j * 64 + t];
    out[t] = fmaxf(o, 0.f);
  }
}

// ---------------- launcher ----------------
extern "C" void kernel_launch(void* const* d_in, const int* in_sizes, int n_in,
                              void* d_out, int out_size, void* d_ws, size_t ws_size,
                              hipStream_t stream) {
  const float* x      = (const float*)d_in[0];
  const int*   ei     = (const int*)  d_in[1];
  const float* eattr  = (const float*)d_in[2];
  const float* enc_w  = (const float*)d_in[3];
  const float* enc_g  = (const float*)d_in[5];
  const float* enc_bt = (const float*)d_in[6];
  const float* edge_w = (const float*)d_in[7];
  const float* edge_b = (const float*)d_in[8];
  const float* W[3]    = {(const float*)d_in[9],  (const float*)d_in[17], (const float*)d_in[25]};
  const float* AS[3]   = {(const float*)d_in[11], (const float*)d_in[19], (const float*)d_in[27]};
  const float* AD[3]   = {(const float*)d_in[12], (const float*)d_in[20], (const float*)d_in[28]};
  const float* AE[3]   = {(const float*)d_in[13], (const float*)d_in[21], (const float*)d_in[29]};
  const float* LEW[3]  = {(const float*)d_in[14], (const float*)d_in[22], (const float*)d_in[30]};
  const float* G[3]    = {(const float*)d_in[15], (const float*)d_in[23], (const float*)d_in[31]};
  const float* BT[3]   = {(const float*)d_in[16], (const float*)d_in[24], (const float*)d_in[32]};
  const float* ro_w1 = (const float*)d_in[33];
  const float* ro_b1 = (const float*)d_in[34];
  const float* ro_w2 = (const float*)d_in[35];
  const float* ro_b2 = (const float*)d_in[36];

  char* ws = (char*)d_ws;
  float*    hA    = (float*)(ws + 0);            // [NN,512]
  float*    hB    = (float*)(ws + 20480000);     // [NN,512] (agg target / scratch)
  f16*      h16   = (f16*)  (ws + 40960000);     // [NN,512] f16
  f16*      w16   = (f16*)  (ws + 51200000);     // [512,512] f16
  float*    xs    = (float*)(ws + 51724288);     // [NN,512]
  float*    sS    = (float*)(ws + 72204288);     // [NN,4]
  float*    sD    = (float*)(ws + 72364288);
  float*    alpha = (float*)(ws + 72524288);     // [NET,4]
  unsigned* mx    = (unsigned*)(ws + 77804288);  // [NN,4] ordered-uint keys
  float*    dn    = (float*)(ws + 77964288);     // [NN,4]
  float*    stats = (float*)(ws + 78124288);     // [2,512]
  float*    sml   = (float*)(ws + 78128384);     // colmean[8] ce[512] edvec[24] ebias[4] selfs[4]
  float* colmean = sml;
  float* ce      = sml + 8;
  float* edvec   = sml + 8 + 512;
  float* ebias   = sml + 8 + 512 + 24;
  float* selfs   = sml + 8 + 512 + 24 + 4;

  float* outNode = (float*)d_out;              // [NN,64]
  float* outGE   = outNode + (long)NN * OUTC;  // [128]
  float* outGF   = outGE + 2 * OUTC;           // [64]

  // edge_attr column means (for self-loop edge features)
  k_fill<<<1, 32, 0, stream>>>(colmean, 0.f, 8);
  k_colmean_accum<<<256, 256, 0, stream>>>(eattr, colmean);
  k_cm_final<<<1, 32, 0, stream>>>(colmean);

  // node encoder: z -> BN -> ReLU -> hA (f32) + h16
  k_enc_mm<<<cdiv((long)NN * HIDC, 256), 256, 0, stream>>>(x, enc_w, hB);
  k_bn_stats<<<HIDC, 256, 0, stream>>>(hB, stats, NN, HIDC);
  k_bn_apply<<<cdiv((long)NN * HIDC, 256), 256, 0, stream>>>(hB, stats, enc_g, enc_bt,
                                                             hA, h16, NN, HIDC, 0);

  const int Kd[3] = {128, 512, 512};
  const int Hh[3] = {4, 4, 1};
  const int Cc[3] = {128, 128, 64};

  for (int l = 0; l < 3; l++) {
    int H = Hh[l], C = Cc[l], HC = H * C, K = Kd[l];
    // weights -> f16, then xs = h16 @ w16 via WMMA (compile-time shapes)
    k_f32_to_f16<<<cdiv((long)K * HC, 256), 256, 0, stream>>>(W[l], w16, (long)K * HC);
    dim3 gg(cdiv(NN, 64), HC / 64);
    if (l == 0)      k_gemm_f16<128, 512><<<gg, 256, 0, stream>>>(h16, w16, xs, NN);
    else if (l == 1) k_gemm_f16<512, 512><<<gg, 256, 0, stream>>>(h16, w16, xs, NN);
    else             k_gemm_f16<512, 64 ><<<gg, 256, 0, stream>>>(h16, w16, xs, NN);
    // per-node attention scores
    k_node_scores<<<cdiv(NN * H, 128), 128, 0, stream>>>(xs, AS[l], AD[l], sS, sD, NN, H, C);
    // folded edge-feature scoring vectors
    k_ce<<<cdiv(HIDC * H, 128), 128, 0, stream>>>(LEW[l], AE[l], ce, H, C);
    k_edge_vec<<<1, 64, 0, stream>>>(edge_w, edge_b, colmean, ce, edvec, ebias, selfs, H);
    // init mx (key 0 == -inf), dn, agg target
    k_fill<<<cdiv((long)NN * NHEADS * 2, 256), 256, 0, stream>>>((float*)mx, 0.f, (long)NN * NHEADS * 2);
    k_fill<<<cdiv((long)NN * HC, 256), 256, 0, stream>>>(hB, 0.f, (long)NN * HC);
    // softmax over incoming edges + aggregation
    k_edge_pass1<<<cdiv(NET, 256), 256, 0, stream>>>(ei, eattr, edvec, ebias, selfs, sS, sD, alpha, mx, H);
    k_edge_pass2<<<cdiv(NET, 256), 256, 0, stream>>>(ei, alpha, mx, dn, H);
    k_agg<<<NET, 128, 0, stream>>>(ei, alpha, dn, xs, hB, H, C);
    // BN + ELU (bias cancels under BN)
    k_bn_stats<<<HC, 256, 0, stream>>>(hB, stats, NN, HC);
    float* of = (l < 2) ? hA : outNode;
    f16*   oh = (l < 2) ? h16 : (f16*)nullptr;
    k_bn_apply<<<cdiv((long)NN * HC, 256), 256, 0, stream>>>(hB, stats, G[l], BT[l], of, oh, NN, HC, 1);
  }

  // readout: mean||max pool -> MLP
  k_pool<<<OUTC, 256, 0, stream>>>(outNode, outGE);
  k_mlp<<<1, 128, 0, stream>>>(outGE, ro_w1, ro_b1, ro_w2, ro_b2, outGF);
}
